// MACHAttention_9663676416534
// MI455X (gfx1250) — compile-verified, matrix operand order checked by reading, not run on hardware
//
#include <hip/hip_runtime.h>
#include <hip/hip_bf16.h>
#include <math.h>

// Problem constants (match reference)
#define BB   4
#define TT   4096
#define DD   1024
#define HH   16
#define DIMh 64
#define LL   8
#define RR   16
#define SSb  128
#define NNh  (BB*HH)     // 64 heads total
#define MMr  (BB*TT)     // 16384 token rows

typedef __attribute__((ext_vector_type(16))) __bf16 v16bf;
typedef __attribute__((ext_vector_type(8)))  __bf16 v8bf;
typedef __attribute__((ext_vector_type(8)))  float  v8f;

union Frag16 { v16bf v; v8bf h[2]; };

// ---------------------------------------------------------------------------
// f32 -> bf16 conversion
// ---------------------------------------------------------------------------
__global__ __launch_bounds__(256) void cvt_f32_bf16(const float* __restrict__ src,
                                                    __bf16* __restrict__ dst, int n) {
    int i = blockIdx.x * blockDim.x + threadIdx.x;
    if (i < n) dst[i] = (__bf16)src[i];
}

// ---------------------------------------------------------------------------
// WMMA bf16 GEMM: out = A[M,1024] @ W[1024,1024]^T  (W row-major [n][k])
// mode 0: store split-head f32 [N=B*H, T, 64]
// mode 1: store plain f32 [M,1024] + bias
// Wave tile: 32(M) x 64(N) = 2x4 WMMA tiles, K step 32.
// ---------------------------------------------------------------------------
__device__ inline v16bf load_frag_bf16(const __bf16* __restrict__ base,
                                       int row, int k0, int half) {
    Frag16 f;
    const __bf16* p = base + (size_t)row * DD + k0 + half * 8;
    f.h[0] = *(const v8bf*)(p);        // K = k0 + half*8 .. +7      (VGPR 0-3)
    f.h[1] = *(const v8bf*)(p + 16);   // K = k0 + 16 + half*8 .. +7 (VGPR 4-7)
    return f.v;
}

__global__ __launch_bounds__(256) void wmma_gemm_bf16(const __bf16* __restrict__ A,
                                                      const __bf16* __restrict__ W,
                                                      const float* __restrict__ bias,
                                                      float* __restrict__ out,
                                                      int M, int mode) {
    const int wid  = (blockIdx.x * blockDim.x + threadIdx.x) >> 5; // global wave id
    const int lane = threadIdx.x & 31;
    const int half = lane >> 4;
    const int lrow = lane & 15;
    const int WN   = DD / 64;                 // 16 wave tiles across N
    const int tm   = (wid / WN) * 32;
    const int tn   = (wid % WN) * 64;
    if (tm >= M) return;

    v8f acc[2][4] = {};

    for (int k0 = 0; k0 < DD; k0 += 32) {
        if (k0 + 32 < DD) {
            __builtin_prefetch(A + (size_t)(tm + lrow) * DD + k0 + 32, 0, 0);
            __builtin_prefetch(W + (size_t)(tn + lrow) * DD + k0 + 32, 0, 0);
        }
        v16bf a[2], b[4];
        a[0] = load_frag_bf16(A, tm + lrow,      k0, half);
        a[1] = load_frag_bf16(A, tm + 16 + lrow, k0, half);
        b[0] = load_frag_bf16(W, tn + lrow,      k0, half);
        b[1] = load_frag_bf16(W, tn + 16 + lrow, k0, half);
        b[2] = load_frag_bf16(W, tn + 32 + lrow, k0, half);
        b[3] = load_frag_bf16(W, tn + 48 + lrow, k0, half);
        #pragma unroll
        for (int i = 0; i < 2; i++)
            #pragma unroll
            for (int j = 0; j < 4; j++)
                acc[i][j] = __builtin_amdgcn_wmma_f32_16x16x32_bf16(
                    false, a[i], false, b[j], (short)0, acc[i][j], false, false);
    }

    #pragma unroll
    for (int i = 0; i < 2; i++) {
        #pragma unroll
        for (int j = 0; j < 4; j++) {
            #pragma unroll
            for (int e = 0; e < 8; e++) {
                int row = tm + i * 16 + half * 8 + e;   // M index
                int col = tn + j * 16 + lrow;           // N index
                float v = acc[i][j][e];
                if (mode == 0) {
                    int bb  = row >> 12;        // / T
                    int t   = row & (TT - 1);
                    int h   = col >> 6;         // / DIMh
                    int dd2 = col & 63;
                    out[(((size_t)(bb * HH + h)) * TT + t) * DIMh + dd2] = v;
                } else {
                    out[(size_t)row * DD + col] = v + bias[col];
                }
            }
        }
    }
}

// ---------------------------------------------------------------------------
// Routing: per (head n, token t): proj = K . planes^T (32 dots of 64),
// tanh/scale, corner logits (L=8 groups x R=16), softmax, top-2 mask,
// renormalize -> probs [N,T,128]
// ---------------------------------------------------------------------------
__global__ __launch_bounds__(256) void routing_kernel(const float* __restrict__ Kbuf,
                                                      const float* __restrict__ planes,
                                                      const float* __restrict__ logit_temp,
                                                      float* __restrict__ probs) {
    __shared__ float sPl[32 * DIMh];
    for (int i = threadIdx.x; i < 32 * DIMh; i += blockDim.x) sPl[i] = planes[i];
    __syncthreads();

    int idx = blockIdx.x * blockDim.x + threadIdx.x;   // over N*T
    int n = idx >> 12;          // / T
    int t = idx & (TT - 1);

    const float* kv = Kbuf + ((size_t)n * TT + t) * DIMh;
    float proj[32];
    #pragma unroll
    for (int p = 0; p < 32; p++) proj[p] = 0.f;
    for (int d = 0; d < DIMh; d++) {
        float kd = kv[d];
        #pragma unroll
        for (int p = 0; p < 32; p++) proj[p] += kd * sPl[p * DIMh + d];
    }

    float scale = fminf(fmaxf(__expf(logit_temp[0]), 0.01f), 20.f);
    float inv = 1.f / scale;
    float* op = probs + ((size_t)n * TT + t) * SSb;

    for (int l = 0; l < LL; l++) {
        float th0 = tanhf(proj[l * 4 + 0]) * inv;
        float th1 = tanhf(proj[l * 4 + 1]) * inv;
        float th2 = tanhf(proj[l * 4 + 2]) * inv;
        float th3 = tanhf(proj[l * 4 + 3]) * inv;
        float lg[RR];
        #pragma unroll
        for (int r = 0; r < RR; r++) {
            // corner sign of bit i for corner r (MSB-first bit order)
            lg[r] = ((r & 8) ? th0 : -th0) + ((r & 4) ? th1 : -th1) +
                    ((r & 2) ? th2 : -th2) + ((r & 1) ? th3 : -th3);
        }
        float mx = lg[0];
        #pragma unroll
        for (int r = 1; r < RR; r++) mx = fmaxf(mx, lg[r]);
        float ex[RR]; float sum = 0.f;
        #pragma unroll
        for (int r = 0; r < RR; r++) { ex[r] = __expf(lg[r] - mx); sum += ex[r]; }
        // top-2 on probs == top-2 on ex
        int i1 = 0; float v1 = ex[0];
        #pragma unroll
        for (int r = 1; r < RR; r++) if (ex[r] > v1) { v1 = ex[r]; i1 = r; }
        int i2 = (i1 == 0) ? 1 : 0; float v2 = ex[i2];
        #pragma unroll
        for (int r = 0; r < RR; r++)
            if (r != i1 && ex[r] > v2) { v2 = ex[r]; i2 = r; }
        // out = (ex[r]/sum) / ((v1+v2)/sum + 1e-6) for kept entries
        float wfac = 1.f / (((v1 + v2) / sum) + 1e-6f) / sum;
        #pragma unroll
        for (int r = 0; r < RR; r++)
            op[l * RR + r] = (r == i1 || r == i2) ? ex[r] * wfac : 0.f;
    }
}

// ---------------------------------------------------------------------------
// Bucket aggregation, LDS-tiled:
//   bucket_K[n,s,:] = sum_t probs[n,t,s]*K[n,t,:] / (A[n,s]+eps)   (same for V)
// One block per (head n, 32-bucket s-tile). t processed in 32-token chunks
// staged through LDS, so K/V are read only SSb/SCH = 4 times each instead
// of 128 times (was ~16 GB of L2 traffic, now ~0.65 GB).
// Thread map: tid = s_local*8 + d_group; each thread owns 8 d-values.
// ---------------------------------------------------------------------------
#define TCH 32   // token chunk
#define SCH 32   // buckets per block
__global__ __launch_bounds__(256) void bucket_agg(const float* __restrict__ Kbuf,
                                                  const float* __restrict__ Vbuf,
                                                  const float* __restrict__ probs,
                                                  float* __restrict__ bK,
                                                  float* __restrict__ bV) {
    __shared__ float sKc[TCH][DIMh];   // 8 KB
    __shared__ float sVc[TCH][DIMh];   // 8 KB
    __shared__ float sPc[TCH][SCH];    // 4 KB
    const int n  = blockIdx.x;          // 64 heads
    const int s0 = blockIdx.y * SCH;    // 4 s-tiles
    const int sl = threadIdx.x >> 3;    // 0..31 bucket within tile
    const int d0 = (threadIdx.x & 7) * 8;

    float accK[8] = {}, accV[8] = {};
    float accA = 0.f;

    for (int t0 = 0; t0 < TT; t0 += TCH) {
        __syncthreads();
        for (int i = threadIdx.x; i < TCH * DIMh; i += 256) {
            int tt = i >> 6, dd = i & 63;
            sKc[tt][dd] = Kbuf[((size_t)n * TT + t0 + tt) * DIMh + dd];
            sVc[tt][dd] = Vbuf[((size_t)n * TT + t0 + tt) * DIMh + dd];
        }
        for (int i = threadIdx.x; i < TCH * SCH; i += 256) {
            int tt = i >> 5, ss = i & 31;
            sPc[tt][ss] = probs[((size_t)n * TT + t0 + tt) * SSb + s0 + ss];
        }
        __syncthreads();
        #pragma unroll 4
        for (int tt = 0; tt < TCH; tt++) {
            float p = sPc[tt][sl];
            accA += p;
            #pragma unroll
            for (int j = 0; j < 8; j++) {
                accK[j] += p * sKc[tt][d0 + j];
                accV[j] += p * sVc[tt][d0 + j];
            }
        }
    }

    float invA = 1.f / (accA + 1e-6f);
    int s = s0 + sl;
    #pragma unroll
    for (int j = 0; j < 8; j++) {
        bK[((size_t)n * SSb + s) * DIMh + d0 + j] = accK[j] * invA;
        bV[((size_t)n * SSb + s) * DIMh + d0 + j] = accV[j] * invA;
    }
}

// ---------------------------------------------------------------------------
// Attention over buckets, online softmax; bucket K/V staged in LDS (64 KB).
// One block = one head x 256 tokens. Output written bf16 in [B,T,D] layout
// (feeds the WMMA output projection directly).
// ---------------------------------------------------------------------------
__global__ __launch_bounds__(256) void bucket_attn(const float* __restrict__ Qbuf,
                                                   const float* __restrict__ bK,
                                                   const float* __restrict__ bV,
                                                   __bf16* __restrict__ attn) {
    __shared__ float sK[SSb * DIMh];
    __shared__ float sV[SSb * DIMh];
    int n = blockIdx.x >> 4;
    int chunk = blockIdx.x & 15;
    for (int i = threadIdx.x; i < SSb * DIMh; i += 256) {
        sK[i] = bK[(size_t)n * SSb * DIMh + i];
        sV[i] = bV[(size_t)n * SSb * DIMh + i];
    }
    __syncthreads();

    int t = chunk * 256 + threadIdx.x;
    const float* qp = Qbuf + ((size_t)n * TT + t) * DIMh;
    float q[DIMh], o[DIMh];
    #pragma unroll
    for (int d = 0; d < DIMh; d++) { q[d] = qp[d]; o[d] = 0.f; }

    float m = -3.0e38f, ssum = 0.f;
    for (int s = 0; s < SSb; s++) {
        float sc = 0.f;
        #pragma unroll
        for (int d = 0; d < DIMh; d++) sc += q[d] * sK[s * DIMh + d];
        sc *= 0.125f;                      // 1/sqrt(64)
        float mn = fmaxf(m, sc);
        float corr = __expf(m - mn);
        float e = __expf(sc - mn);
        ssum = ssum * corr + e;
        #pragma unroll
        for (int d = 0; d < DIMh; d++) o[d] = o[d] * corr + e * sV[s * DIMh + d];
        m = mn;
    }
    float invs = 1.f / ssum;
    int bb = n >> 4;          // / H
    int h  = n & 15;
    __bf16* op = attn + ((size_t)(bb * TT + t)) * DD + h * DIMh;
    #pragma unroll
    for (int d = 0; d < DIMh; d++) op[d] = (__bf16)(o[d] * invs);
}

// ---------------------------------------------------------------------------
// host side
// ---------------------------------------------------------------------------
extern "C" void kernel_launch(void* const* d_in, const int* in_sizes, int n_in,
                              void* d_out, int out_size, void* d_ws, size_t ws_size,
                              hipStream_t stream) {
    const float* x          = (const float*)d_in[0];
    const float* planes_T   = (const float*)d_in[1];
    /* d_in[2] protos_T: corners computed analytically */
    const float* Wq         = (const float*)d_in[3];
    const float* Wk         = (const float*)d_in[4];
    const float* Wv         = (const float*)d_in[5];
    const float* Wout       = (const float*)d_in[6];
    const float* b_out      = (const float*)d_in[7];
    const float* logit_temp = (const float*)d_in[8];
    float* out = (float*)d_out;

    uint8_t* ws = (uint8_t*)d_ws;
    size_t off = 0;
    auto alloc = [&](size_t bytes) -> void* {
        void* p = ws + off;
        off += (bytes + 255) & ~(size_t)255;
        return p;
    };

    __bf16* xb   = (__bf16*)alloc((size_t)MMr * DD * sizeof(__bf16));
    __bf16* wqb  = (__bf16*)alloc((size_t)DD * DD * sizeof(__bf16));
    __bf16* wkb  = (__bf16*)alloc((size_t)DD * DD * sizeof(__bf16));
    __bf16* wvb  = (__bf16*)alloc((size_t)DD * DD * sizeof(__bf16));
    __bf16* wob  = (__bf16*)alloc((size_t)DD * DD * sizeof(__bf16));
    float*  Qb   = (float*)alloc((size_t)NNh * TT * DIMh * sizeof(float));
    float*  Kb   = (float*)alloc((size_t)NNh * TT * DIMh * sizeof(float));
    float*  Vb   = (float*)alloc((size_t)NNh * TT * DIMh * sizeof(float));
    float*  prb  = (float*)alloc((size_t)NNh * TT * SSb * sizeof(float));
    float*  bK   = (float*)alloc((size_t)NNh * SSb * DIMh * sizeof(float));
    float*  bV   = (float*)alloc((size_t)NNh * SSb * DIMh * sizeof(float));
    __bf16* attn = (__bf16*)alloc((size_t)MMr * DD * sizeof(__bf16));

    // 1) convert activations + weights to bf16
    {
        int n = MMr * DD;
        cvt_f32_bf16<<<(n + 255) / 256, 256, 0, stream>>>(x, xb, n);
        int nw = DD * DD;
        cvt_f32_bf16<<<(nw + 255) / 256, 256, 0, stream>>>(Wq, wqb, nw);
        cvt_f32_bf16<<<(nw + 255) / 256, 256, 0, stream>>>(Wk, wkb, nw);
        cvt_f32_bf16<<<(nw + 255) / 256, 256, 0, stream>>>(Wv, wvb, nw);
        cvt_f32_bf16<<<(nw + 255) / 256, 256, 0, stream>>>(Wout, wob, nw);
    }

    // 2) QKV projections via WMMA, stored split-head [N,T,64] f32
    int gemm_waves  = (MMr / 32) * (DD / 64);      // 8192
    int gemm_blocks = gemm_waves * 32 / 256;       // 1024
    wmma_gemm_bf16<<<gemm_blocks, 256, 0, stream>>>(xb, wqb, nullptr, Qb, MMr, 0);
    wmma_gemm_bf16<<<gemm_blocks, 256, 0, stream>>>(xb, wkb, nullptr, Kb, MMr, 0);
    wmma_gemm_bf16<<<gemm_blocks, 256, 0, stream>>>(xb, wvb, nullptr, Vb, MMr, 0);

    // 3) bucket routing probabilities
    routing_kernel<<<NNh * TT / 256, 256, 0, stream>>>(Kb, planes_T, logit_temp, prb);

    // 4) bucket aggregation + normalization (LDS-tiled)
    bucket_agg<<<dim3(NNh, SSb / SCH), 256, 0, stream>>>(Kb, Vb, prb, bK, bV);

    // 5) attention over buckets (LDS-staged K/V, online softmax), bf16 out
    bucket_attn<<<NNh * (TT / 256), 256, 0, stream>>>(Qb, bK, bV, attn);

    // 6) output projection + bias via WMMA, f32 to d_out
    wmma_gemm_bf16<<<gemm_blocks, 256, 0, stream>>>(attn, wob, b_out, out, MMr, 1);
}
